// CausalSelfAttention_35974646071793
// MI455X (gfx1250) — matrix operand order repaired and checked by reading, not compiled, not run on hardware
//
#include <hip/hip_runtime.h>

typedef _Float16 half_t;
typedef __attribute__((ext_vector_type(16))) _Float16 v16h;
typedef __attribute__((ext_vector_type(8)))  _Float16 v8h;
typedef __attribute__((ext_vector_type(8)))  float    v8f;
typedef __attribute__((ext_vector_type(4)))  float    v4f;

// ---------------------------------------------------------------------------
// WMMA helper: D = A(16x32 f16) * B(32x16 f16) + C(16x16 f32)
// ---------------------------------------------------------------------------
__device__ __forceinline__ v8f wmma_f16(v16h a, v16h b, v8f c) {
  return __builtin_amdgcn_wmma_f32_16x16x32_f16(
      /*neg_a=*/false, a, /*neg_b=*/false, b,
      /*c_mod=*/(short)0, c, /*reuse_a=*/false, /*reuse_b=*/false);
}

// A-operand fragment (16x32 f16):
// lane L: M = L%16; halves[h] = K = 16*(L/16) + h.
// p points at (row0,col0), ld = row stride in halves.
__device__ __forceinline__ v16h load_a_frag(const half_t* p, int ld) {
  const int lid = threadIdx.x & 31;
  const half_t* row = p + (lid & 15) * ld;
  const int kb = (lid < 16) ? 0 : 16;
  v8h lo = *(const v8h*)(row + kb);
  v8h hi = *(const v8h*)(row + kb + 8);
  union { v16h v; v8h h[2]; } u;
  u.h[0] = lo; u.h[1] = hi;
  return u.v;
}

// B-operand fragment (32x16 f16), contiguous-k layout (same k order as load_a_frag):
// lane L: N = L%16; halves[h] = K = 16*(L/16) + h.
// p points at a [N][K] (column-of-B major) tile, ld = stride in halves.
__device__ __forceinline__ v16h load_b_frag(const half_t* p, int ld) {
  const int lid = threadIdx.x & 31;
  const half_t* row = p + (lid & 15) * ld + ((lid >> 4) << 4);
  return *(const v16h*)row;  // 32 contiguous bytes
}

// ---------------------------------------------------------------------------
// Generic WMMA GEMM:  Cmat[M,N] = A[M,K] * W[K,N] + bias[N]
//   EPI==0: scatter epilogue into Q/K (head-major f16) and V-transposed f16
//   EPI==1: plain fp32 row-major store
// Block: 256 threads (8 wave32), tile 128x128, K-chunk 32.
// ---------------------------------------------------------------------------
template<typename AT, int EPI>
__global__ __launch_bounds__(256) void wmma_gemm_kernel(
    const AT* __restrict__ A, const float* __restrict__ W,
    const float* __restrict__ bias,
    half_t* __restrict__ qout, half_t* __restrict__ kout,
    half_t* __restrict__ vtout, float* __restrict__ fout,
    int M, int N, int K)
{
  constexpr int LDA = 40;   // halves; 16B-aligned rows for v8h loads
  constexpr int LDB = 48;   // halves; 32B-aligned rows for v16h loads
  __shared__ half_t As[128 * LDA];
  __shared__ half_t Bs[128 * LDB];

  const int t   = threadIdx.x;
  const int lid = t & 31;
  const int w   = t >> 5;
  const int wm  = w >> 2;          // 0..1  (64-row slab)
  const int wn  = w & 3;           // 0..3  (32-col slab)
  const int bn  = blockIdx.x;
  const int bm  = blockIdx.y;

  const v8f zero = {0.f,0.f,0.f,0.f,0.f,0.f,0.f,0.f};
  v8f acc[4][2];
  #pragma unroll
  for (int i = 0; i < 4; i++)
    #pragma unroll
    for (int j = 0; j < 2; j++) acc[i][j] = zero;

  for (int k0 = 0; k0 < K; k0 += 32) {
    // Stage A tile 128x32 (convert -> f16)
    #pragma unroll
    for (int i = 0; i < 4; i++) {
      int idx = t + i * 256;
      int r = idx >> 3, c4 = (idx & 7) << 2;
      const AT* src = A + (size_t)(bm * 128 + r) * K + k0 + c4;
      half_t* dst = &As[r * LDA + c4];
      #pragma unroll
      for (int jj = 0; jj < 4; jj++) dst[jj] = (half_t)src[jj];
    }
    // Stage W tile transposed: Bs[n][kk] = W[k0+kk][bn*128+n] (f32 -> f16)
    #pragma unroll
    for (int i = 0; i < 4; i++) {
      int idx = t + i * 256;
      int kk = idx >> 5, n4 = (idx & 31) << 2;
      const float* src = W + (size_t)(k0 + kk) * N + bn * 128 + n4;
      v4f wv = *(const v4f*)src;
      #pragma unroll
      for (int jj = 0; jj < 4; jj++) Bs[(n4 + jj) * LDB + kk] = (half_t)wv[jj];
    }
    if (k0 + 32 < K) {  // CDNA5 global_prefetch_b8 for next K tile
      __builtin_prefetch(A + (size_t)(bm * 128 + (t >> 3)) * K + k0 + 32, 0, 1);
      __builtin_prefetch(W + (size_t)(k0 + 32 + (t >> 5)) * N + bn * 128, 0, 1);
    }
    __syncthreads();

    v16h af[4], bf[2];
    #pragma unroll
    for (int i = 0; i < 4; i++)
      af[i] = load_a_frag(&As[(wm * 64 + i * 16) * LDA], LDA);
    #pragma unroll
    for (int j = 0; j < 2; j++)
      bf[j] = load_b_frag(&Bs[(wn * 32 + j * 16) * LDB], LDB);
    #pragma unroll
    for (int i = 0; i < 4; i++)
      #pragma unroll
      for (int j = 0; j < 2; j++)
        acc[i][j] = wmma_f16(af[i], bf[j], acc[i][j]);
    __syncthreads();
  }

  // Epilogue.  C-layout: elem e of acc = row (e + 8*(lid/16)), col = lid%16.
  const int coln  = lid & 15;
  const int rhalf = (lid >> 4) << 3;
  #pragma unroll
  for (int i = 0; i < 4; i++) {
    #pragma unroll
    for (int j = 0; j < 2; j++) {
      const int gcol  = bn * 128 + wn * 32 + j * 16 + coln;
      const int growb = bm * 128 + wm * 64 + i * 16 + rhalf;
      const float bv  = bias[gcol];
      if (EPI == 0) {
        // gcol in [0, 3C): split into q/k/v, per-head layouts
        const int sel = gcol >> 10;        // 0=q 1=k 2=v   (C=1024)
        const int c   = gcol & 1023;
        const int h   = c >> 6;            // head        (D=64)
        const int d   = c & 63;
        #pragma unroll
        for (int e = 0; e < 8; e++) {
          const int grow = growb + e;
          const int bidx = grow >> 11;     // batch      (T=2048)
          const int tt   = grow & 2047;
          const int bh   = (bidx << 4) + h;
          const half_t hv = (half_t)(acc[i][j][e] + bv);
          if (sel == 0)      qout[((size_t)bh * 2048 + tt) * 64 + d] = hv;
          else if (sel == 1) kout[((size_t)bh * 2048 + tt) * 64 + d] = hv;
          else               vtout[((size_t)bh * 64 + d) * 2048 + tt] = hv;
        }
      } else {
        #pragma unroll
        for (int e = 0; e < 8; e++)
          fout[(size_t)(growb + e) * N + gcol] = acc[i][j][e] + bv;
      }
    }
  }
}

// ---------------------------------------------------------------------------
// Flash attention (causal, online softmax).  One wave owns 16 query rows.
// Q[bh][t][d], K[bh][t][d] f16;  Vt[bh][d][t] f16 (pre-transposed);
// Y[b*T+t][h*64+d] f16 output (feeds the proj GEMM as its A operand).
// ---------------------------------------------------------------------------
__global__ __launch_bounds__(256) void flash_attn_kernel(
    const half_t* __restrict__ Q, const half_t* __restrict__ Kh,
    const half_t* __restrict__ Vt, half_t* __restrict__ Y)
{
  constexpr int T = 2048, D = 64, C = 1024;
  constexpr int LDP = 40;
  __shared__ half_t Plds[8 * 16 * LDP];   // per-wave P transpose scratch

  const int tid   = threadIdx.x;
  const int lid   = tid & 31;
  const int w     = tid >> 5;
  const int bh    = blockIdx.y;
  const int qbase = blockIdx.x * 128 + w * 16;

  const half_t* Qp = Q  + (size_t)bh * T * D;
  const half_t* Kp = Kh + (size_t)bh * T * D;
  const half_t* Vp = Vt + (size_t)bh * D * T;

  const int coln  = lid & 15;
  const int rhalf = (lid >> 4) << 3;   // row offset for C-layout
  const int khalf = (lid >> 4) << 4;   // K offset for B-layout

  // Q A-operand fragments for d=0..31 and d=32..63
  const v16h aq0 = load_a_frag(Qp + (size_t)qbase * D, D);
  const v16h aq1 = load_a_frag(Qp + (size_t)qbase * D + 32, D);

  const v8f zero = {0.f,0.f,0.f,0.f,0.f,0.f,0.f,0.f};
  v8f acc[4] = {zero, zero, zero, zero};
  float mrow[8], lrow[8];
  #pragma unroll
  for (int e = 0; e < 8; e++) { mrow[e] = -3.0e38f; lrow[e] = 0.f; }

  half_t* Pw = &Plds[w * 16 * LDP];

  // Causal: keys <= qbase+15 only -> skip future blocks entirely.
  for (int j0 = 0; j0 < qbase + 16; j0 += 32) {
    // ---- S = Q * K^T over 32 keys (two 16-key N tiles) ----
    v8f s[2];
    #pragma unroll
    for (int nt = 0; nt < 2; nt++) {
      const half_t* kb = Kp + (size_t)(j0 + nt * 16) * D;
      v16h b0 = load_b_frag(kb, D);        // d = 0..31
      v16h b1 = load_b_frag(kb + 32, D);   // d = 32..63
      s[nt] = wmma_f16(aq1, b1, wmma_f16(aq0, b0, zero));
    }
    // ---- scale + causal mask ----
    #pragma unroll
    for (int nt = 0; nt < 2; nt++)
      #pragma unroll
      for (int e = 0; e < 8; e++) {
        const float sv  = s[nt][e] * 0.125f;   // 1/sqrt(64)
        const int kcol  = j0 + nt * 16 + coln;
        const int qrow  = qbase + rhalf + e;
        s[nt][e] = (kcol <= qrow) ? sv : -3.0e38f;
      }
    // ---- online softmax stats (reduce across the 16-lane column group) ----
    #pragma unroll
    for (int e = 0; e < 8; e++) {
      float v = fmaxf(s[0][e], s[1][e]);
      v = fmaxf(v, __shfl_xor(v, 1, 32));
      v = fmaxf(v, __shfl_xor(v, 2, 32));
      v = fmaxf(v, __shfl_xor(v, 4, 32));
      v = fmaxf(v, __shfl_xor(v, 8, 32));
      const float mnew  = fmaxf(mrow[e], v);
      const float alpha = __expf(mrow[e] - mnew);
      mrow[e] = mnew;
      const float p0 = __expf(s[0][e] - mnew);
      const float p1 = __expf(s[1][e] - mnew);
      s[0][e] = p0; s[1][e] = p1;
      float rs = p0 + p1;
      rs += __shfl_xor(rs, 1, 32);
      rs += __shfl_xor(rs, 2, 32);
      rs += __shfl_xor(rs, 4, 32);
      rs += __shfl_xor(rs, 8, 32);
      lrow[e] = lrow[e] * alpha + rs;
      #pragma unroll
      for (int dt = 0; dt < 4; dt++) acc[dt][e] *= alpha;
    }
    // ---- transpose P (C-layout -> A-layout) via per-wave LDS ----
    #pragma unroll
    for (int nt = 0; nt < 2; nt++)
      #pragma unroll
      for (int e = 0; e < 8; e++)
        Pw[(rhalf + e) * LDP + nt * 16 + coln] = (half_t)s[nt][e];
    asm volatile("s_wait_dscnt 0" ::: "memory");   // wave-local RAW on LDS
    const v16h pa = load_a_frag(Pw, LDP);
    // ---- O += P * V ;  Vt gives contiguous B-operand fragments ----
    #pragma unroll
    for (int dt = 0; dt < 4; dt++) {
      const half_t* vp = Vp + (size_t)(dt * 16 + coln) * T + j0 + khalf;
      const v16h vb = *(const v16h*)vp;
      acc[dt] = wmma_f16(pa, vb, acc[dt]);
    }
  }

  // ---- normalize and store Y[b*T+t][h*64+d] as f16 ----
  const int b = bh >> 4, h = bh & 15;
  #pragma unroll
  for (int dt = 0; dt < 4; dt++)
    #pragma unroll
    for (int e = 0; e < 8; e++) {
      const int tq = qbase + rhalf + e;
      const float yv = acc[dt][e] / lrow[e];
      Y[(size_t)(b * T + tq) * C + h * 64 + dt * 16 + coln] = (half_t)yv;
    }
}

// ---------------------------------------------------------------------------
// Host launcher
// ---------------------------------------------------------------------------
extern "C" void kernel_launch(void* const* d_in, const int* in_sizes, int n_in,
                              void* d_out, int out_size, void* d_ws, size_t ws_size,
                              hipStream_t stream) {
  (void)in_sizes; (void)n_in; (void)out_size; (void)ws_size;
  const float* x      = (const float*)d_in[0];
  const float* W_attn = (const float*)d_in[1];
  const float* b_attn = (const float*)d_in[2];
  const float* W_proj = (const float*)d_in[3];
  const float* b_proj = (const float*)d_in[4];
  float* out = (float*)d_out;

  constexpr int Bq = 4, T = 2048, C = 1024, H = 16, D = 64;
  constexpr int M = Bq * T;                       // 8192
  constexpr size_t headElems = (size_t)Bq * H * T * D;  // 8.39M f16 each

  half_t* Qh = (half_t*)d_ws;
  half_t* Kh = Qh + headElems;
  half_t* Vt = Kh + headElems;
  half_t* Yh = Vt + headElems;                    // [M][C] f16

  dim3 blk(256);
  // 1) QKV GEMM + bias, scatter to head-major Q/K and transposed V
  wmma_gemm_kernel<float, 0><<<dim3(3 * C / 128, M / 128), blk, 0, stream>>>(
      x, W_attn, b_attn, Qh, Kh, Vt, nullptr, M, 3 * C, C);
  // 2) causal flash attention
  flash_attn_kernel<<<dim3(T / 128, Bq * H), blk, 0, stream>>>(Qh, Kh, Vt, Yh);
  // 3) output projection + bias -> fp32
  wmma_gemm_kernel<half_t, 1><<<dim3(C / 128, M / 128), blk, 0, stream>>>(
      Yh, W_proj, b_proj, nullptr, nullptr, nullptr, out, M, C, C);
}
